// LlamaMLPInfer_80290118632063
// MI455X (gfx1250) — matrix operand order repaired and checked by reading, not compiled, not run on hardware
//
#include <hip/hip_runtime.h>

// ---------------------------------------------------------------------------
// LlamaMLPInfer (SwiGLU, int4-group-quant) for MI455X / gfx1250.
// Pre-dequant weights int32->f16 (+x f32->f16) into ws, then two
// f16 WMMA GEMMs: 128Mx64N WG tiles, 8 waves (4Mx2N), LDS double buffering,
// register-pipelined staging, one barrier per 32-K step. Inner loop manually
// unrolled x2 (ping-pong) and pinned with unroll(disable) to keep the
// mainloop compact (I$-friendly). Grid M-fastest so each N-stripe's f16
// weights are L2-shared by its 32 M-blocks; x_h / inter stay L2-hot.
// ---------------------------------------------------------------------------

#define HDIM 4096
#define IDIM 14336
#define TOK  4096
#define GRP  128
#define AST  40      // LDS row stride (f16): 32 data + 8 pad; 80B, 16B-aligned

typedef _Float16 v16h __attribute__((ext_vector_type(16)));
typedef _Float16 v8h  __attribute__((ext_vector_type(8)));
typedef float    v8f  __attribute__((ext_vector_type(8)));
typedef float    v4f  __attribute__((ext_vector_type(4)));
typedef int      v4i  __attribute__((ext_vector_type(4)));

static __device__ __forceinline__ v16h cat16(v8h lo, v8h hi) {
  return __builtin_shufflevector(lo, hi, 0, 1, 2, 3, 4, 5, 6, 7,
                                 8, 9, 10, 11, 12, 13, 14, 15);
}
static __device__ __forceinline__ v8f wmma_f16(v16h a, v16h b, v8f c) {
  return __builtin_amdgcn_wmma_f32_16x16x32_f16(false, a, false, b,
                                                (short)0, c, false, false);
}

// ---------------------------------------------------------------------------
// Prep kernels (elementwise, HBM-bound)
// ---------------------------------------------------------------------------
__global__ __launch_bounds__(256)
void k_cvt_x(const float* __restrict__ x, _Float16* __restrict__ xh) {
  const size_t i = ((size_t)blockIdx.x * 256 + threadIdx.x) * 8;
  v4f a = *(const v4f*)(x + i);
  v4f b = *(const v4f*)(x + i + 4);
  v8h h;
  h[0] = (_Float16)a[0]; h[1] = (_Float16)a[1];
  h[2] = (_Float16)a[2]; h[3] = (_Float16)a[3];
  h[4] = (_Float16)b[0]; h[5] = (_Float16)b[1];
  h[6] = (_Float16)b[2]; h[7] = (_Float16)b[3];
  *(v8h*)(xh + i) = h;
}

__global__ __launch_bounds__(256)
void k_dequant_w(const int* __restrict__ q, const float* __restrict__ s,
                 _Float16* __restrict__ wh, int ld) {
  const int row = blockIdx.y;
  const int k   = (blockIdx.x * 256 + threadIdx.x) * 8;   // 8 elts, one group
  const size_t off = (size_t)row * ld + k;
  v4i q0 = *(const v4i*)(q + off);
  v4i q1 = *(const v4i*)(q + off + 4);
  const float sc = s[(size_t)row * (ld >> 7) + (k >> 7)];
  v8h h;
#pragma unroll
  for (int j = 0; j < 4; ++j) {
    h[j]     = (_Float16)((float)q0[j] * sc);
    h[4 + j] = (_Float16)((float)q1[j] * sc);
  }
  *(v8h*)(wh + off) = h;
}

// ---------------------------------------------------------------------------
// MMA step helpers (A/B fragment reads from LDS + wmma; B frags die fast)
// ---------------------------------------------------------------------------
static __device__ __forceinline__ void mma_step_gu(
    const _Float16* lA, int offA0, int offB0,
    v8f& g00, v8f& g01, v8f& g10, v8f& g11,
    v8f& u00, v8f& u01, v8f& u10, v8f& u11) {
  const _Float16* pa0 = lA + offA0;
  const _Float16* pa1 = pa0 + 16 * AST;
  v16h a0 = cat16(*(const v8h*)pa0, *(const v8h*)(pa0 + 8));
  v16h a1 = cat16(*(const v8h*)pa1, *(const v8h*)(pa1 + 8));

  const _Float16* pg0 = lA + 128 * AST + offB0;
  {
    v16h bg0 = cat16(*(const v8h*)pg0, *(const v8h*)(pg0 + 8));
    g00 = wmma_f16(a0, bg0, g00);
    g10 = wmma_f16(a1, bg0, g10);
  }
  {
    const _Float16* pg1 = pg0 + 16 * AST;
    v16h bg1 = cat16(*(const v8h*)pg1, *(const v8h*)(pg1 + 8));
    g01 = wmma_f16(a0, bg1, g01);
    g11 = wmma_f16(a1, bg1, g11);
  }
  {
    const _Float16* pu0 = pg0 + 64 * AST;
    v16h bu0 = cat16(*(const v8h*)pu0, *(const v8h*)(pu0 + 8));
    u00 = wmma_f16(a0, bu0, u00);
    u10 = wmma_f16(a1, bu0, u10);
  }
  {
    const _Float16* pu1 = pg0 + 80 * AST;
    v16h bu1 = cat16(*(const v8h*)pu1, *(const v8h*)(pu1 + 8));
    u01 = wmma_f16(a0, bu1, u01);
    u11 = wmma_f16(a1, bu1, u11);
  }
}

static __device__ __forceinline__ void mma_step_d(
    const _Float16* lA, int offA0, int offB0,
    v8f& d00, v8f& d01, v8f& d10, v8f& d11) {
  const _Float16* pa0 = lA + offA0;
  const _Float16* pa1 = pa0 + 16 * AST;
  v16h a0 = cat16(*(const v8h*)pa0, *(const v8h*)(pa0 + 8));
  v16h a1 = cat16(*(const v8h*)pa1, *(const v8h*)(pa1 + 8));

  const _Float16* pb0 = lA + 128 * AST + offB0;
  {
    v16h b0 = cat16(*(const v8h*)pb0, *(const v8h*)(pb0 + 8));
    d00 = wmma_f16(a0, b0, d00);
    d10 = wmma_f16(a1, b0, d10);
  }
  {
    const _Float16* pb1 = pb0 + 16 * AST;
    v16h b1 = cat16(*(const v8h*)pb1, *(const v8h*)(pb1 + 8));
    d01 = wmma_f16(a0, b1, d01);
    d11 = wmma_f16(a1, b1, d11);
  }
}

// One pipelined K-step: load next tile to regs, MMA on current LDS buffer,
// store next tile to the other LDS buffer, barrier.
static __device__ __forceinline__ void gu_body(
    const _Float16*& pA, const _Float16*& pB,
    _Float16* ad, _Float16* bd, const _Float16* lcur, int offA0, int offB0,
    v8f& g00, v8f& g01, v8f& g10, v8f& g11,
    v8f& u00, v8f& u01, v8f& u10, v8f& u11) {
  v8h ra0 = *(const v8h*)(pA);
  v8h ra1 = *(const v8h*)(pA + 8);
  v8h rb0 = *(const v8h*)(pB);
  v8h rb1 = *(const v8h*)(pB + 8);
  pA += 32; pB += 32;
  mma_step_gu(lcur, offA0, offB0, g00, g01, g10, g11, u00, u01, u10, u11);
  *(v8h*)ad = ra0; *(v8h*)(ad + 8) = ra1;
  *(v8h*)bd = rb0; *(v8h*)(bd + 8) = rb1;
  __syncthreads();
}

static __device__ __forceinline__ void d_body(
    const _Float16*& pA, const _Float16*& pB,
    _Float16* ad, _Float16* bd, const _Float16* lcur, int offA0, int offB0,
    v8f& d00, v8f& d01, v8f& d10, v8f& d11) {
  v8h ra0 = *(const v8h*)(pA);
  v8h ra1 = *(const v8h*)(pA + 8);
  v8h rb0 = *(const v8h*)(pB);
  pA += 32; pB += 32;
  mma_step_d(lcur, offA0, offB0, d00, d01, d10, d11);
  *(v8h*)ad = ra0; *(v8h*)(ad + 8) = ra1;
  *(v8h*)bd = rb0;
  __syncthreads();
}

// ---------------------------------------------------------------------------
// Kernel 1: inter[t,i] = silu(x@Wg^T) * (x@Wu^T), f16.
// ---------------------------------------------------------------------------
__global__ __launch_bounds__(256)
void k_gateup(const _Float16* __restrict__ xh,
              const _Float16* __restrict__ wgh,
              const _Float16* __restrict__ wuh,
              _Float16* __restrict__ inter) {
  __shared__ _Float16 lds[2][(128 + 64 + 64) * AST];   // A | Bgate | Bup

  const int tid  = threadIdx.x;
  const int lane = tid & 31;
  const int w    = tid >> 5;
  const int mw   = w >> 1;                 // 0..3
  const int nw   = w & 1;                  // 0..1

  const int rowWG = blockIdx.x << 7;       // M (fastest -> N-stripe bands)
  const int colWG = blockIdx.y << 6;       // N over IDIM

  // cooperative staging: A = 128x32 (16 f16/thr), B = 2 x 64x32 (16 f16/thr)
  const int aRow = tid >> 1;
  const int aOff = (tid & 1) << 4;
  const int bMat = tid >> 7;               // 0 gate, 1 up
  const int bN   = (tid & 127) >> 1;
  const int bOff = (tid & 1) << 4;

  const _Float16* pA = xh + (size_t)(rowWG + aRow) * HDIM + aOff;
  const _Float16* pB = (bMat ? wuh : wgh) + (size_t)(colWG + bN) * HDIM + bOff;
  _Float16* aD0 = &lds[0][aRow * AST + aOff];
  _Float16* aD1 = &lds[1][aRow * AST + aOff];
  _Float16* bD0 = &lds[0][(128 + bMat * 64 + bN) * AST + bOff];
  _Float16* bD1 = &lds[1][(128 + bMat * 64 + bN) * AST + bOff];

  v8f g00 = {}, g01 = {}, g10 = {}, g11 = {};
  v8f u00 = {}, u01 = {}, u10 = {}, u11 = {};

  // prologue: stage step 0 into buffer 0
  {
    v8h ra0 = *(const v8h*)(pA);
    v8h ra1 = *(const v8h*)(pA + 8);
    v8h rb0 = *(const v8h*)(pB);
    v8h rb1 = *(const v8h*)(pB + 8);
    *(v8h*)aD0 = ra0; *(v8h*)(aD0 + 8) = ra1;
    *(v8h*)bD0 = rb0; *(v8h*)(bD0 + 8) = rb1;
    pA += 32; pB += 32;
  }
  __syncthreads();

  const int mB    = mw << 5, nB = nw << 5;
  const int offA0 = (mB + (lane & 15)) * AST + ((lane >> 4) << 4);
  const int offB0 = (nB + (lane & 15)) * AST + ((lane >> 4) << 4);

  const int STEPS = HDIM / 32;             // 128
#pragma clang loop unroll(disable)
  for (int it = 0; it < (STEPS - 2) / 2; ++it) {   // steps 0..STEPS-3
    gu_body(pA, pB, aD1, bD1, lds[0], offA0, offB0,
            g00, g01, g10, g11, u00, u01, u10, u11);
    gu_body(pA, pB, aD0, bD0, lds[1], offA0, offB0,
            g00, g01, g10, g11, u00, u01, u10, u11);
  }
  gu_body(pA, pB, aD1, bD1, lds[0], offA0, offB0,    // step STEPS-2 (even)
          g00, g01, g10, g11, u00, u01, u10, u11);
  mma_step_gu(lds[1], offA0, offB0,                  // step STEPS-1 (odd)
              g00, g01, g10, g11, u00, u01, u10, u11);

  // epilogue: silu(g)*u -> f16. C layout: VGPR r -> M=r+8*(lane/16), N=lane%16
  const int em = (lane >> 4) << 3;
  const int en = lane & 15;
#pragma unroll
  for (int r = 0; r < 8; ++r) {
    const int m0 = rowWG + mB + em + r, m1 = m0 + 16;
    const int c0 = colWG + nB + en,     c1 = c0 + 16;
    float g, a;
    g = g00[r]; a = (g / (1.0f + __expf(-g))) * u00[r];
    inter[(size_t)m0 * IDIM + c0] = (_Float16)a;
    g = g01[r]; a = (g / (1.0f + __expf(-g))) * u01[r];
    inter[(size_t)m0 * IDIM + c1] = (_Float16)a;
    g = g10[r]; a = (g / (1.0f + __expf(-g))) * u10[r];
    inter[(size_t)m1 * IDIM + c0] = (_Float16)a;
    g = g11[r]; a = (g / (1.0f + __expf(-g))) * u11[r];
    inter[(size_t)m1 * IDIM + c1] = (_Float16)a;
  }
}

// ---------------------------------------------------------------------------
// Kernel 2: out[t,h] = inter @ Wd^T, f32 out.
// ---------------------------------------------------------------------------
__global__ __launch_bounds__(256)
void k_down(const _Float16* __restrict__ inter,
            const _Float16* __restrict__ wdh,
            float* __restrict__ out) {
  __shared__ _Float16 lds[2][(128 + 64) * AST];        // A | Bdown

  const int tid  = threadIdx.x;
  const int lane = tid & 31;
  const int w    = tid >> 5;
  const int mw   = w >> 1;
  const int nw   = w & 1;

  const int rowWG = blockIdx.x << 7;
  const int colWG = blockIdx.y << 6;                   // N over HDIM

  const int aRow = tid >> 1;
  const int aOff = (tid & 1) << 4;
  const int bN   = tid >> 2;                           // 0..63
  const int bOff = (tid & 3) << 3;                     // 8 f16 per thread

  const _Float16* pA = inter + (size_t)(rowWG + aRow) * IDIM + aOff;
  const _Float16* pB = wdh + (size_t)(colWG + bN) * IDIM + bOff;
  _Float16* aD0 = &lds[0][aRow * AST + aOff];
  _Float16* aD1 = &lds[1][aRow * AST + aOff];
  _Float16* bD0 = &lds[0][(128 + bN) * AST + bOff];
  _Float16* bD1 = &lds[1][(128 + bN) * AST + bOff];

  v8f d00 = {}, d01 = {}, d10 = {}, d11 = {};

  {
    v8h ra0 = *(const v8h*)(pA);
    v8h ra1 = *(const v8h*)(pA + 8);
    v8h rb0 = *(const v8h*)(pB);
    *(v8h*)aD0 = ra0; *(v8h*)(aD0 + 8) = ra1;
    *(v8h*)bD0 = rb0;
    pA += 32; pB += 32;
  }
  __syncthreads();

  const int mB    = mw << 5, nB = nw << 5;
  const int offA0 = (mB + (lane & 15)) * AST + ((lane >> 4) << 4);
  const int offB0 = (nB + (lane & 15)) * AST + ((lane >> 4) << 4);

  const int STEPS = IDIM / 32;             // 448
#pragma clang loop unroll(disable)
  for (int it = 0; it < (STEPS - 2) / 2; ++it) {
    d_body(pA, pB, aD1, bD1, lds[0], offA0, offB0, d00, d01, d10, d11);
    d_body(pA, pB, aD0, bD0, lds[1], offA0, offB0, d00, d01, d10, d11);
  }
  d_body(pA, pB, aD1, bD1, lds[0], offA0, offB0, d00, d01, d10, d11);
  mma_step_d(lds[1], offA0, offB0, d00, d01, d10, d11);

  const int em = (lane >> 4) << 3;
  const int en = lane & 15;
#pragma unroll
  for (int r = 0; r < 8; ++r) {
    const int m0 = rowWG + mB + em + r, m1 = m0 + 16;
    const int c0 = colWG + nB + en,     c1 = c0 + 16;
    out[(size_t)m0 * HDIM + c0] = d00[r];
    out[(size_t)m0 * HDIM + c1] = d01[r];
    out[(size_t)m1 * HDIM + c0] = d10[r];
    out[(size_t)m1 * HDIM + c1] = d11[r];
  }
}

// ---------------------------------------------------------------------------
extern "C" void kernel_launch(void* const* d_in, const int* in_sizes, int n_in,
                              void* d_out, int out_size, void* d_ws,
                              size_t ws_size, hipStream_t stream) {
  (void)in_sizes; (void)n_in; (void)out_size; (void)ws_size;
  const float* x   = (const float*)d_in[0];
  const int*   gq  = (const int*)d_in[1];
  const float* gs  = (const float*)d_in[2];
  const int*   uq  = (const int*)d_in[3];
  const float* us  = (const float*)d_in[4];
  const int*   dq  = (const int*)d_in[5];
  const float* dsc = (const float*)d_in[6];

  char* ws = (char*)d_ws;   // layout (bytes):
  _Float16* xh    = (_Float16*)(ws);                      //  33,554,432
  _Float16* wgh   = (_Float16*)(ws + 33554432ull);        // 117,440,512
  _Float16* wuh   = (_Float16*)(ws + 150994944ull);       // 117,440,512
  _Float16* wdh   = (_Float16*)(ws + 268435456ull);       // 117,440,512
  _Float16* inter = (_Float16*)(ws + 385875968ull);       // 117,440,512
  float*    out   = (float*)d_out;

  dim3 blk(256, 1, 1);
  k_cvt_x<<<dim3((TOK * HDIM) / 2048, 1, 1), blk, 0, stream>>>(x, xh);
  k_dequant_w<<<dim3(HDIM / 2048, IDIM, 1), blk, 0, stream>>>(gq, gs, wgh, HDIM);
  k_dequant_w<<<dim3(HDIM / 2048, IDIM, 1), blk, 0, stream>>>(uq, us, wuh, HDIM);
  k_dequant_w<<<dim3(IDIM / 2048, HDIM, 1), blk, 0, stream>>>(dq, dsc, wdh, IDIM);

  k_gateup<<<dim3(TOK / 128, IDIM / 64, 1), blk, 0, stream>>>(xh, wgh, wuh, inter);
  k_down  <<<dim3(TOK / 128, HDIM / 64, 1), blk, 0, stream>>>(inter, wdh, out);
}
